// Indexer_10599979286546
// MI455X (gfx1250) — matrix operand order repaired and checked by reading, not compiled, not run on hardware
//
#include <hip/hip_runtime.h>
#include <hip/hip_bf16.h>

// ---------------------------------------------------------------------------
// Types for CDNA5 WMMA (gfx1250, wave32)
// ---------------------------------------------------------------------------
typedef __attribute__((ext_vector_type(16))) __bf16 v16bf;
typedef __attribute__((ext_vector_type(8)))  __bf16 v8bf;
typedef __attribute__((ext_vector_type(8)))  float  v8f;
typedef __attribute__((ext_vector_type(4)))  float  v4f;
typedef __attribute__((ext_vector_type(4)))  int    v4i;

#define AS1 __attribute__((address_space(1)))
#define AS3 __attribute__((address_space(3)))

union BF16x16 { v16bf v; v8bf h[2]; };

__device__ __forceinline__ v8f wmma_bf16(v16bf a, v16bf b, v8f c) {
    // D = A(16x32 bf16) * B(32x16 bf16) + C(16x16 f32)
    return __builtin_amdgcn_wmma_f32_16x16x32_bf16(
        /*neg_a=*/false, a, /*neg_b=*/false, b,
        /*c_mod=*/(short)0, c, /*reuse_a=*/false, /*reuse_b=*/false);
}

// Bit-exact f32 <-> bf16 (round-to-nearest-even), no reliance on __bf16 ALU ops
__device__ __forceinline__ __bf16 f2bf(float f) {
    unsigned u = __builtin_bit_cast(unsigned, f);
    u += 0x7FFFu + ((u >> 16) & 1u);
    unsigned short h = (unsigned short)(u >> 16);
    return __builtin_bit_cast(__bf16, h);
}
__device__ __forceinline__ float bf2f(__bf16 h) {
    unsigned short b = __builtin_bit_cast(unsigned short, h);
    unsigned u = ((unsigned)b) << 16;
    return __builtin_bit_cast(float, u);
}
__device__ __forceinline__ v8f v8f_zero() {
    v8f z = {0.f,0.f,0.f,0.f,0.f,0.f,0.f,0.f};
    return z;
}

#define S_TOK   2048
#define DIM     2048
#define NH      64
#define HD      128
#define QLR     1536
#define NHD     (NH*HD)      // 8192
#define HAD_SCALE 0.08838834764831845f          // 128^-0.5
#define W_SCALE   0.011048543456039806f         // 64^-0.5 * 128^-0.5

// ---------------------------------------------------------------------------
// Kernel 1: per-token K path (x@wk^T -> layernorm -> RoPE -> Hadamard -> bf16)
//           plus per-token head weights w = (x@wproj^T)*scale
// ---------------------------------------------------------------------------
__global__ __launch_bounds__(128) void kprep_kernel(
    const float* __restrict__ x, const float* __restrict__ wk,
    const float* __restrict__ knw, const float* __restrict__ knb,
    const float* __restrict__ wproj,
    const float* __restrict__ cosT, const float* __restrict__ sinT,
    __bf16* __restrict__ Kbf, float* __restrict__ W)
{
    __shared__ float xs[DIM];
    __shared__ float red[HD];
    const int t = blockIdx.x, tid = threadIdx.x;

    #pragma unroll
    for (int j = 0; j < 16; ++j)
        xs[tid + 128*j] = x[(size_t)t*DIM + tid + 128*j];
    __syncthreads();

    // k_d = x_row . wk[d,:]
    float kd = 0.f;
    {
        const v4f* wr4 = (const v4f*)(wk + (size_t)tid*DIM);
        const v4f* xs4 = (const v4f*)xs;
        float a = 0.f;
        #pragma unroll 8
        for (int i = 0; i < DIM/4; ++i) {
            v4f xa = xs4[i], wb = wr4[i];
            a = fmaf(xa[0], wb[0], a); a = fmaf(xa[1], wb[1], a);
            a = fmaf(xa[2], wb[2], a); a = fmaf(xa[3], wb[3], a);
        }
        kd = a;
    }

    // head weights (threads 0..63)
    if (tid < NH) {
        const v4f* wr4 = (const v4f*)(wproj + (size_t)tid*DIM);
        const v4f* xs4 = (const v4f*)xs;
        float a = 0.f;
        #pragma unroll 8
        for (int i = 0; i < DIM/4; ++i) {
            v4f xa = xs4[i], wb = wr4[i];
            a = fmaf(xa[0], wb[0], a); a = fmaf(xa[1], wb[1], a);
            a = fmaf(xa[2], wb[2], a); a = fmaf(xa[3], wb[3], a);
        }
        W[(size_t)t*NH + tid] = a * W_SCALE;
    }

    // layernorm over 128 dims
    red[tid] = kd; __syncthreads();
    for (int off = 64; off >= 1; off >>= 1) {
        if (tid < off) red[tid] += red[tid + off];
        __syncthreads();
    }
    const float mu = red[0] * (1.f/128.f);
    __syncthreads();
    const float d0 = kd - mu;
    red[tid] = d0 * d0; __syncthreads();
    for (int off = 64; off >= 1; off >>= 1) {
        if (tid < off) red[tid] += red[tid + off];
        __syncthreads();
    }
    const float var = red[0] * (1.f/128.f);
    __syncthreads();
    kd = d0 * rsqrtf(var + 1e-6f) * knw[tid] + knb[tid];

    // RoPE (rotate-half) on first 64 dims, position t
    red[tid] = kd; __syncthreads();
    float nk = kd;
    if (tid < 64) {
        const int j = tid & 31;
        const float c = cosT[(size_t)t*32 + j], s = sinT[(size_t)t*32 + j];
        nk = (tid < 32) ? (red[tid]*c - red[tid+32]*s)
                        : (red[tid-32]*s + red[tid]*c);
    }
    __syncthreads();
    kd = nk;

    // Hadamard transform (FWHT), then scale
    #pragma unroll
    for (int b = 1; b < 128; b <<= 1) {
        red[tid] = kd; __syncthreads();
        const float o = red[tid ^ b];
        kd = (tid & b) ? (o - kd) : (kd + o);
        __syncthreads();
    }
    Kbf[(size_t)t*HD + tid] = f2bf(kd * HAD_SCALE);
}

// ---------------------------------------------------------------------------
// Kernel 2: f32 -> bf16 conversion (grid-stride)
// ---------------------------------------------------------------------------
__global__ void cvt_kernel(const float* __restrict__ src, __bf16* __restrict__ dst, int n)
{
    int i = blockIdx.x * blockDim.x + threadIdx.x;
    const int stride = gridDim.x * blockDim.x;
    for (; i < n; i += stride) dst[i] = f2bf(src[i]);
}

// ---------------------------------------------------------------------------
// Kernel 3: Q = qr @ wq_b^T  (bf16 WMMA, f32 accumulate)
// Both operands K-major -> fragments load directly per ISA bf16 layouts.
// grid (NHD/128, S/16), block 256 (8 waves, wave -> one 16-col n-tile)
// ---------------------------------------------------------------------------
__global__ __launch_bounds__(256) void qproj_kernel(
    const __bf16* __restrict__ qrb, const __bf16* __restrict__ wqb,
    __bf16* __restrict__ Q)
{
    const int lane = threadIdx.x & 31, wave = threadIdx.x >> 5;
    const int lm = lane & 15, hi = lane >> 4;
    const int sbase = blockIdx.y * 16;
    const int nbase = blockIdx.x * 128 + wave * 16;

    const __bf16* arow = qrb + (size_t)(sbase + lm) * QLR;
    const __bf16* brow = wqb + (size_t)(nbase + lm) * QLR;

    v8f c = v8f_zero();
    #pragma unroll 4
    for (int kc = 0; kc < QLR/32; ++kc) {
        BF16x16 a, b;
        // A 16x32 bf16: lane lm = row M; two 8-elem K runs selected by hi
        a.h[0] = *(const v8bf*)(arow + kc*32 + hi*8);
        a.h[1] = *(const v8bf*)(arow + kc*32 + 16 + hi*8);
        // B 32x16 bf16: lane lm = col N; two 8-elem K runs selected by hi
        b.h[0] = *(const v8bf*)(brow + kc*32 + hi*8);
        b.h[1] = *(const v8bf*)(brow + kc*32 + 16 + hi*8);
        c = wmma_bf16(a.v, b.v, c);
    }
    // C layout: VGPR i -> M = i + 8*hi, N = lm
    #pragma unroll
    for (int i = 0; i < 8; ++i) {
        const int M = i + hi*8;
        Q[(size_t)(sbase + M) * NHD + nbase + lm] = f2bf(c[i]);
    }
}

// ---------------------------------------------------------------------------
// Kernel 4: in-place per-head RoPE + Hadamard on Q (bf16)
// ---------------------------------------------------------------------------
__global__ __launch_bounds__(128) void qpost_kernel(
    __bf16* __restrict__ Q, const float* __restrict__ cosT, const float* __restrict__ sinT)
{
    __shared__ float v[HD];
    const int s = blockIdx.x, tid = threadIdx.x;
    float cv = 0.f, sv = 0.f;
    if (tid < 64) {
        const int j = tid & 31;
        cv = cosT[(size_t)s*32 + j];
        sv = sinT[(size_t)s*32 + j];
    }
    for (int h = 0; h < NH; ++h) {
        __bf16* q = Q + (size_t)s*NHD + h*HD;
        float qd = bf2f(q[tid]);
        v[tid] = qd; __syncthreads();
        float nq = qd;
        if (tid < 64)
            nq = (tid < 32) ? (v[tid]*cv - v[tid+32]*sv)
                            : (v[tid-32]*sv + v[tid]*cv);
        __syncthreads();
        qd = nq;
        #pragma unroll
        for (int b = 1; b < 128; b <<= 1) {
            v[tid] = qd; __syncthreads();
            const float o = v[tid ^ b];
            qd = (tid & b) ? (o - qd) : (qd + o);
            __syncthreads();
        }
        q[tid] = f2bf(qd * HAD_SCALE);
    }
}

// ---------------------------------------------------------------------------
// Kernel 5: index_score[s,t] = sum_h w[s,h]*relu(q[s,h,:].k[t,:]) + mask
// Transposed tiles: M = t (K-matrix rows as WMMA A), N = s (Q cols as WMMA B)
// so w[s,h] is lane-uniform. Q half-tile staged via CDNA5 async direct-to-LDS
// (GLOBAL_LOAD_ASYNC_TO_LDS_B128, ASYNCcnt), padded LDS pitch for bank spread.
// grid (S/256, S/16), block 256; dynamic LDS 135680 B (2 workgroups/WGP).
// ---------------------------------------------------------------------------
#define SC_P 4104   // LDS row pitch (bf16 elems): 32*128 + 8 pad

__global__ __launch_bounds__(256) void scores_kernel(
    const __bf16* __restrict__ Q, const __bf16* __restrict__ Kb,
    const float* __restrict__ W, float* __restrict__ outScore)
{
    extern __shared__ char smem[];
    __bf16* qs = (__bf16*)smem;                         // 16 * SC_P * 2 bytes
    float*  w2 = (float*)(smem + 16 * SC_P * 2);        // [64][17] transposed

    const int tid = threadIdx.x;
    const int lane = tid & 31, wave = tid >> 5;
    const int lm = lane & 15, hi = lane >> 4;
    const int sbase = blockIdx.y * 16;
    const int tb0   = blockIdx.x * 256 + wave * 32;

    // stage head-weights transposed (w2[h][srow]) — conflict-free reads
    #pragma unroll
    for (int e = tid; e < 16*NH; e += 256) {
        const int r = e >> 6, h = e & 63;
        w2[h*17 + r] = W[(size_t)(sbase + r)*NH + h];
    }

    v8f acc[2]; acc[0] = v8f_zero(); acc[1] = v8f_zero();

    for (int half = 0; half < 2; ++half) {
        __syncthreads();
        // stage q[sbase..+16][half*32 heads * 128] -> LDS via async copy (128KB)
        {
            const int r = tid >> 4, seg = tid & 15;
            const __bf16* src = Q + (size_t)(sbase + r)*NHD + half*4096 + seg*256;
            __bf16* dst = qs + r*SC_P + seg*256;
            #pragma unroll
            for (int j = 0; j < 32; ++j) {
                __builtin_amdgcn_global_load_async_to_lds_b128(
                    (AS1 v4i*)(src + j*8),
                    (AS3 v4i*)(dst + j*8),
                    /*offset=*/0, /*cpol=*/0);
            }
        }
        __builtin_amdgcn_s_wait_asynccnt(0);
        __syncthreads();

        #pragma unroll
        for (int tile = 0; tile < 2; ++tile) {
            const int tbase = tb0 + tile*16;
            // A fragments: K-matrix rows (M = t), 4 chunks of K=32 over HD=128
            BF16x16 a[4];
            const __bf16* krow = Kb + (size_t)(tbase + lm)*HD;
            #pragma unroll
            for (int ch = 0; ch < 4; ++ch) {
                a[ch].h[0] = *(const v8bf*)(krow + ch*32 + hi*8);
                a[ch].h[1] = *(const v8bf*)(krow + ch*32 + 16 + hi*8);
            }
            #pragma unroll 1
            for (int hh = 0; hh < 32; ++hh) {
                const __bf16* qb = qs + lm*SC_P + hh*HD + hi*8;
                v8f tacc = v8f_zero();
                #pragma unroll
                for (int ch = 0; ch < 4; ++ch) {
                    BF16x16 b;
                    b.h[0] = *(const v8bf*)(qb + ch*32);
                    b.h[1] = *(const v8bf*)(qb + ch*32 + 16);
                    tacc = wmma_bf16(a[ch].v, b.v, tacc);
                }
                const float wsc = w2[(half*32 + hh)*17 + lm];  // lane-uniform in N
                #pragma unroll
                for (int i = 0; i < 8; ++i) {
                    float r = tacc[i];
                    r = r > 0.f ? r : 0.f;
                    acc[tile][i] = fmaf(wsc, r, acc[tile][i]);
                }
            }
        }
    }

    // store transposed (C: M=t, N=s) with causal mask applied inline
    const int s_idx = sbase + lm;
    #pragma unroll
    for (int tile = 0; tile < 2; ++tile) {
        const int tbase = tb0 + tile*16;
        #pragma unroll
        for (int i = 0; i < 8; ++i) {
            const int t_idx = tbase + i + hi*8;
            const float v = acc[tile][i] + (t_idx > s_idx ? -1e9f : 0.f);
            outScore[(size_t)s_idx * S_TOK + t_idx] = v;
        }
    }
}

// ---------------------------------------------------------------------------
// Kernel 6: per-row full descending argsort (TOPK == S) via bitonic sort of
// packed u64 keys: (~monotone(score) << 32) | index  -> stable top_k order.
// ---------------------------------------------------------------------------
__global__ __launch_bounds__(256) void sort_kernel(
    const float* __restrict__ score, int* __restrict__ outIdx)
{
    __shared__ unsigned long long keys[S_TOK];
    const int s = blockIdx.x, tid = threadIdx.x;

    #pragma unroll
    for (int j = 0; j < 8; ++j) {
        const int i = tid + j*256;
        const float f = score[(size_t)s*S_TOK + i];
        unsigned u = __builtin_bit_cast(unsigned, f);
        u ^= (u & 0x80000000u) ? 0xFFFFFFFFu : 0x80000000u;  // monotone ascending
        u = ~u;                                               // descending score
        keys[i] = (((unsigned long long)u) << 32) | (unsigned)i;
    }
    __syncthreads();

    for (int k = 2; k <= S_TOK; k <<= 1) {
        for (int j = k >> 1; j > 0; j >>= 1) {
            #pragma unroll
            for (int m = 0; m < 8; ++m) {
                const int i = tid + m*256;
                const int l = i ^ j;
                if (l > i) {
                    const bool up = (i & k) == 0;
                    const unsigned long long A = keys[i], B = keys[l];
                    const bool sw = up ? (A > B) : (A < B);
                    if (sw) { keys[i] = B; keys[l] = A; }
                }
            }
            __syncthreads();
        }
    }

    #pragma unroll
    for (int j = 0; j < 8; ++j) {
        const int i = tid + j*256;
        outIdx[(size_t)s*S_TOK + i] = (int)(keys[i] & 0xFFFFFFFFu);
    }
}

// ---------------------------------------------------------------------------
// Host entry
// ---------------------------------------------------------------------------
extern "C" void kernel_launch(void* const* d_in, const int* in_sizes, int n_in,
                              void* d_out, int out_size, void* d_ws, size_t ws_size,
                              hipStream_t stream)
{
    (void)in_sizes; (void)n_in; (void)out_size; (void)ws_size;

    const float* x     = (const float*)d_in[0];   // [1,2048,2048]
    const float* qr    = (const float*)d_in[1];   // [1,2048,1536]
    /* d_in[2] = start_pos (0, unused) */
    const float* cosT  = (const float*)d_in[3];   // [2048,32]
    const float* sinT  = (const float*)d_in[4];   // [2048,32]
    /* d_in[5] = mask (recomputed inline) */
    const float* wq_b  = (const float*)d_in[6];   // [8192,1536]
    const float* wk    = (const float*)d_in[7];   // [128,2048]
    const float* knw   = (const float*)d_in[8];   // [128]
    const float* knb   = (const float*)d_in[9];   // [128]
    const float* wproj = (const float*)d_in[10];  // [64,2048]

    char* ws = (char*)d_ws;
    __bf16* Kbf = (__bf16*)(ws + 0);           //  512 KB  [2048][128]
    float*  W   = (float*) (ws + 524288);      //  512 KB  [2048][64]
    __bf16* qrb = (__bf16*)(ws + 1048576);     //    6 MB  [2048][1536]
    __bf16* wqb = (__bf16*)(ws + 7340032);     //   24 MB  [8192][1536]
    __bf16* Qws = (__bf16*)(ws + 32505856);    //   32 MB  [2048][8192]

    int*   outIdx   = (int*)d_out;                              // 4M int32
    float* outScore = ((float*)d_out) + (size_t)S_TOK * S_TOK;  // 4M f32

    const int scoresLds = 16 * SC_P * 2 + 64 * 17 * 4;  // 135680 B
    (void)hipFuncSetAttribute((const void*)scores_kernel,
                              hipFuncAttributeMaxDynamicSharedMemorySize, scoresLds);

    kprep_kernel<<<S_TOK, 128, 0, stream>>>(x, wk, knw, knb, wproj, cosT, sinT, Kbf, W);
    cvt_kernel<<<2048, 256, 0, stream>>>(qr,   qrb, S_TOK * QLR);
    cvt_kernel<<<4096, 256, 0, stream>>>(wq_b, wqb, NHD * QLR);
    qproj_kernel<<<dim3(NHD/128, S_TOK/16), 256, 0, stream>>>(qrb, wqb, Qws);
    qpost_kernel<<<S_TOK, 128, 0, stream>>>(Qws, cosT, sinT);
    scores_kernel<<<dim3(S_TOK/256, S_TOK/16), 256, scoresLds, stream>>>(Qws, Kbf, W, outScore);
    sort_kernel<<<S_TOK, 256, 0, stream>>>(outScore, outIdx);
}
